// VQEmbedding_17738214933194
// MI455X (gfx1250) — matrix operand order repaired and checked by reading, not compiled, not run on hardware
//
#include <hip/hip_runtime.h>
#include <hip/hip_bf16.h>
#include <math.h>

#define NUM_EMB 8192
#define EMB_DIM 512
#define N_TOK   32768

#define MTILE  32     // token rows per block (2 row-groups x 16, both per wave)
#define NCHUNK 128    // codebook codes per chunk (8 waves x 16 cols)
#define DSUB   64     // dims per LDS weight sub-tile (8 sub-tiles per chunk)

typedef __attribute__((ext_vector_type(8)))  __bf16 v8bf;
typedef __attribute__((ext_vector_type(16))) __bf16 v16bf;
typedef __attribute__((ext_vector_type(8)))  float  v8f;

__device__ __forceinline__ unsigned short f32_to_bf16_rne(float f) {
    union { float f; unsigned int u; } c; c.f = f;
    unsigned int u = c.u;
    u += 0x7FFFu + ((u >> 16) & 1u);   // round-to-nearest-even
    return (unsigned short)(u >> 16);
}

// Async global->LDS 128-bit copy (ASYNCcnt-tracked, bypasses VGPRs).
__device__ __forceinline__ void async_ld_b128(unsigned ldsAddr, unsigned gOff,
                                              const unsigned short* base) {
    asm volatile("global_load_async_to_lds_b128 %0, %1, %2 offset:0"
                 :: "v"(ldsAddr), "v"(gOff), "s"(base) : "memory");
}

// ---------------- f32 -> bf16 conversion (grid-stride) ----------------
__global__ void cvt_bf16_kernel(const float* __restrict__ src,
                                unsigned short* __restrict__ dst, int n) {
    int i = blockIdx.x * blockDim.x + threadIdx.x;
    int stride = gridDim.x * blockDim.x;
    for (; i < n; i += stride) dst[i] = f32_to_bf16_rne(src[i]);
}

// ---------------- 0.5*||w_k||^2, one wave per code ----------------
__global__ void wnorm_kernel(const float* __restrict__ w, float* __restrict__ wnh) {
    int wave = (blockIdx.x * blockDim.x + threadIdx.x) >> 5;
    int lane = threadIdx.x & 31;
    if (wave >= NUM_EMB) return;
    const float* row = w + (size_t)wave * EMB_DIM;
    float s = 0.f;
    for (int j = lane; j < EMB_DIM; j += 32) { float v = row[j]; s += v * v; }
    #pragma unroll
    for (int m = 16; m >= 1; m >>= 1) s += __shfl_xor(s, m, 32);
    if (lane == 0) wnh[wave] = 0.5f * s;
}

// ---------------- main WMMA argmax kernel ----------------
__global__ __launch_bounds__(256) void vq_wmma_kernel(
    const unsigned short* __restrict__ xbf,   // [N_TOK][EMB_DIM] bf16 bits
    const unsigned short* __restrict__ wbf,   // [NUM_EMB][EMB_DIM] bf16 bits
    const float* __restrict__ wnh,            // [NUM_EMB] 0.5*||w||^2
    int* __restrict__ bestIdxOut)             // [N_TOK]
{
    __shared__ unsigned short ldsW[2][NCHUNK][DSUB];   // 2 x 16 KB double buffer
    __shared__ float redVal[2][8][16];
    __shared__ int   redIdx[2][8][16];

    const int tid  = threadIdx.x;
    const int lane = tid & 31;
    const int wv   = tid >> 5;           // wave 0..7 == col-group
    const int nOff = wv << 4;            // 16 cols per wave within 128-code chunk
    const int col  = lane & 15;
    const int hi   = lane >> 4;          // half-wave
    const int aLo  = hi ? 8  : 0;        // 16-bit A layout: 8-elem per-lane chunks
    const int bLo  = hi ? 16 : 0;        // unused; B is loaded with aLo in A's 8-elem chunk order
    const int rowBase = blockIdx.x * MTILE;

    const unsigned ldsWB0 = (unsigned)(uintptr_t)&ldsW[0][0][0];
    const unsigned ldsWB1 = (unsigned)(uintptr_t)&ldsW[1][0][0];

    // ---- prologue: kick off async load of chunk 0 / sub-tile 0 ----
    // sub-tile = 128 codes x 64 dims = 1024 uint4; 8 uint4 per code row
    #pragma unroll
    for (int q = 0; q < 4; ++q) {
        int j = tid + q * 256;
        int code = j >> 3, part = j & 7;
        async_ld_b128(ldsWB0 + (unsigned)j * 16,
                      (unsigned)(code * 1024 + part * 16), wbf);
    }

    // ---- hoist A fragments for BOTH row groups (32 rows x 512 dims) into VGPRs ----
    v16bf afrag[2][16];
    #pragma unroll
    for (int mg = 0; mg < 2; ++mg) {
        const unsigned short* xr = xbf + (size_t)(rowBase + mg * 16 + col) * EMB_DIM;
        #pragma unroll
        for (int g = 0; g < 16; ++g) {
            v8bf a0 = *(const v8bf*)(xr + g * 32 + aLo);
            v8bf a1 = *(const v8bf*)(xr + g * 32 + 16 + aLo);
            afrag[mg][g] = __builtin_shufflevector(a0, a1,
                           0,1,2,3,4,5,6,7,8,9,10,11,12,13,14,15);
        }
    }

    float bestVal[2][8];
    int   bestIdx[2][8];
    #pragma unroll
    for (int mg = 0; mg < 2; ++mg)
        #pragma unroll
        for (int r = 0; r < 8; ++r) { bestVal[mg][r] = -3.4e38f; bestIdx[mg][r] = 0; }

    for (int cb = 0; cb < NUM_EMB; cb += NCHUNK) {
        v8f acc0 = {}, acc1 = {};
        #pragma unroll
        for (int s = 0; s < 8; ++s) {
            // issue async loads for the NEXT sub-tile into the other buffer
            int ncb, ndb;
            bool doIssue;
            if (s < 7) { ncb = cb;          ndb = (s + 1) * DSUB; doIssue = true; }
            else       { ncb = cb + NCHUNK; ndb = 0; doIssue = (ncb < NUM_EMB); }
            if (doIssue) {
                unsigned nbase = (s & 1) ? ldsWB0 : ldsWB1;
                #pragma unroll
                for (int q = 0; q < 4; ++q) {
                    int j = tid + q * 256;
                    int code = j >> 3, part = j & 7;
                    async_ld_b128(nbase + (unsigned)j * 16,
                                  (unsigned)((ncb + code) * 1024 + ndb * 2 + part * 16),
                                  wbf);
                }
                asm volatile("s_wait_asynccnt 0x4" ::: "memory");
            } else {
                asm volatile("s_wait_asynccnt 0x0" ::: "memory");
            }
            __syncthreads();   // current sub-tile visible to all waves

            const unsigned short* wrow = &ldsW[s & 1][nOff + col][0];
            #pragma unroll
            for (int kk = 0; kk < 2; ++kk) {      // 2 K-steps of 32 dims
                v8bf b0 = *(const v8bf*)(wrow + kk * 32 + aLo);
                v8bf b1 = *(const v8bf*)(wrow + kk * 32 + 16 + aLo);
                v16bf b = __builtin_shufflevector(b0, b1,
                           0,1,2,3,4,5,6,7,8,9,10,11,12,13,14,15);
                // one B fragment feeds BOTH row-group WMMAs (2x LDS reuse)
                acc0 = __builtin_amdgcn_wmma_f32_16x16x32_bf16(
                           false, afrag[0][s * 2 + kk], false, b, (short)0, acc0,
                           false, false);
                acc1 = __builtin_amdgcn_wmma_f32_16x16x32_bf16(
                           false, afrag[1][s * 2 + kk], false, b, (short)0, acc1,
                           false, false);
            }
            __syncthreads();   // done reading this buffer; async may overwrite it
        }

        // score = x.w - 0.5||w||^2 ; ascending scan + strict '>' keeps first winner
        const int code = cb + nOff + col;
        const float sub = wnh[code];
        #pragma unroll
        for (int r = 0; r < 8; ++r) {
            float s0 = acc0[r] - sub;
            float s1 = acc1[r] - sub;
            bool b0 = s0 > bestVal[0][r];
            bool b1 = s1 > bestVal[1][r];
            bestVal[0][r] = b0 ? s0 : bestVal[0][r];
            bestIdx[0][r] = b0 ? code : bestIdx[0][r];
            bestVal[1][r] = b1 ? s1 : bestVal[1][r];
            bestIdx[1][r] = b1 ? code : bestIdx[1][r];
        }
    }

    // Butterfly across 16 columns within each half-wave (tie-break: lowest index)
    #pragma unroll
    for (int mg = 0; mg < 2; ++mg) {
        #pragma unroll
        for (int r = 0; r < 8; ++r) {
            float v = bestVal[mg][r]; int i = bestIdx[mg][r];
            #pragma unroll
            for (int m = 1; m < 16; m <<= 1) {
                float ov = __shfl_xor(v, m, 32);
                int   oi = __shfl_xor(i, m, 32);
                bool take = (ov > v) || (ov == v && oi < i);
                v = take ? ov : v;
                i = take ? oi : i;
            }
            bestVal[mg][r] = v; bestIdx[mg][r] = i;
        }
    }
    if (col == 0) {
        #pragma unroll
        for (int mg = 0; mg < 2; ++mg)
            #pragma unroll
            for (int r = 0; r < 8; ++r) {
                redVal[mg][wv][hi * 8 + r] = bestVal[mg][r];
                redIdx[mg][wv][hi * 8 + r] = bestIdx[mg][r];
            }
    }
    __syncthreads();
    if (tid < MTILE) {
        int g = tid >> 4, rr = tid & 15;
        float v = redVal[g][0][rr]; int i = redIdx[g][0][rr];
        #pragma unroll
        for (int n = 1; n < 8; ++n) {
            float ov = redVal[g][n][rr]; int oi = redIdx[g][n][rr];
            bool take = (ov > v) || (ov == v && oi < i);
            v = take ? ov : v; i = take ? oi : i;
        }
        bestIdxOut[rowBase + tid] = i;
    }
}

// ---------------- histogram zero ----------------
__global__ void zero_hist_kernel(int* __restrict__ hist) {
    int i = blockIdx.x * blockDim.x + threadIdx.x;
    if (i < NUM_EMB) hist[i] = 0;
}

// ---------------- gather + index output + histogram ----------------
__global__ void gather_kernel(const int* __restrict__ bestIdx,
                              const float* __restrict__ weight,
                              float* __restrict__ outQ,
                              float* __restrict__ outIdx,
                              int* __restrict__ hist)
{
    int n = blockIdx.x;
    int k = bestIdx[n];
    const float4* src = (const float4*)(weight + (size_t)k * EMB_DIM);
    float4* dst = (float4*)(outQ + (size_t)n * EMB_DIM);
    for (int i = threadIdx.x; i < EMB_DIM / 4; i += blockDim.x) dst[i] = src[i];
    if (threadIdx.x == 0) {
        outIdx[n] = (float)k;
        atomicAdd(&hist[k], 1);
    }
}

// ---------------- perplexity ----------------
__global__ void perplexity_kernel(const int* __restrict__ hist,
                                  float* __restrict__ outPerp)
{
    __shared__ float red[256];
    float s = 0.f;
    for (int i = threadIdx.x; i < NUM_EMB; i += blockDim.x) {
        float p = (float)hist[i] * (1.0f / (float)N_TOK);
        s += p * logf(p + 1e-10f);
    }
    red[threadIdx.x] = s;
    __syncthreads();
    for (int off = 128; off > 0; off >>= 1) {
        if (threadIdx.x < off) red[threadIdx.x] += red[threadIdx.x + off];
        __syncthreads();
    }
    if (threadIdx.x == 0) *outPerp = expf(-red[0]);
}

extern "C" void kernel_launch(void* const* d_in, const int* in_sizes, int n_in,
                              void* d_out, int out_size, void* d_ws, size_t ws_size,
                              hipStream_t stream) {
    const float* x = (const float*)d_in[0];   // [32768,512]
    const float* w = (const float*)d_in[1];   // [8192,512]

    float* out     = (float*)d_out;
    float* outQ    = out;                               // 32768*512
    float* outIdx  = out + (size_t)N_TOK * EMB_DIM;     // 32768
    float* outPerp = outIdx + N_TOK;                    // 1

    char* ws = (char*)d_ws;
    unsigned short* xbf = (unsigned short*)ws;                         // 33,554,432 B
    unsigned short* wbf = (unsigned short*)(ws + 33554432);            //  8,388,608 B
    float* wnh   = (float*)(ws + 33554432 + 8388608);                  //     32,768 B
    int*   bIdx  = (int*)  (ws + 33554432 + 8388608 + 32768);          //    131,072 B
    int*   hist  = (int*)  (ws + 33554432 + 8388608 + 32768 + 131072); //     32,768 B

    cvt_bf16_kernel<<<2048, 256, 0, stream>>>(x, xbf, N_TOK * EMB_DIM);
    cvt_bf16_kernel<<<1024, 256, 0, stream>>>(w, wbf, NUM_EMB * EMB_DIM);
    wnorm_kernel<<<(NUM_EMB * 32) / 256, 256, 0, stream>>>(w, wnh);
    vq_wmma_kernel<<<N_TOK / MTILE, 256, 0, stream>>>(xbf, wbf, wnh, bIdx);
    zero_hist_kernel<<<NUM_EMB / 256, 256, 0, stream>>>(hist);
    gather_kernel<<<N_TOK, 128, 0, stream>>>(bIdx, w, outQ, outIdx, hist);
    perplexity_kernel<<<1, 256, 0, stream>>>(hist, outPerp);
}
